// GATNet_79173427134952
// MI455X (gfx1250) — matrix operand order repaired and checked by reading, not compiled, not run on hardware
//
#include <hip/hip_runtime.h>
#include <hip/hip_bf16.h>

// ---------------------------------------------------------------------------
// GATv2 (2 layers, 4 heads x 32) + down-projection for MI455X (gfx1250).
// Dense projections use v_wmma_f32_16x16x32_f16 (f16 in, f32 accumulate).
// Edge softmax/aggregation: wave32-per-edge coalesced gather + f32 atomics.
// ---------------------------------------------------------------------------

#define HEADS   4
#define DHEAD   32
#define HID     128          // HEADS * DHEAD
#define KDIM    128          // inner dim of every GEMM here
#define NEG_SLOPE 0.2f

typedef __attribute__((ext_vector_type(16))) _Float16 v16h;
typedef __attribute__((ext_vector_type(8)))  _Float16 v8h;
typedef __attribute__((ext_vector_type(8)))  float    v8f;

// ---- helpers ---------------------------------------------------------------

__device__ __forceinline__ unsigned orderedOf(float f) {
  unsigned u = __float_as_uint(f);
  return (u & 0x80000000u) ? ~u : (u | 0x80000000u);   // monotone float->uint
}
__device__ __forceinline__ float unorderedOf(unsigned u) {
  u = (u & 0x80000000u) ? (u & 0x7FFFFFFFu) : ~u;
  return __uint_as_float(u);
}

// ---- f32 -> f16 conversion -------------------------------------------------

__global__ void cvt_f32_to_f16_kernel(const float* __restrict__ src,
                                      _Float16* __restrict__ dst, int n) {
  int i = blockIdx.x * blockDim.x + threadIdx.x;
  if (i < n) dst[i] = (_Float16)src[i];
}

// ---- WMMA GEMM: out[N,OUTD] = A[N,128] @ W[OUTD,128]^T + bias --------------
// One wave per 16x16 output tile. A and W are row-major with K contiguous,
// so both fragments are loaded in the same contiguous K order (K = g*16 + s).
// OUTD is a template constant so the 8 D-stores use immediate offsets.

template <int OUTD>
__global__ void wmma_gemm_bias_kernel(const _Float16* __restrict__ A,    // [N,128]
                                      const _Float16* __restrict__ W,    // [OUTD,128]
                                      const float*    __restrict__ bias, // [OUTD]
                                      float*          __restrict__ out,  // [N,OUTD]
                                      int N) {
  constexpr int colTiles = OUTD >> 4;
  int gtid = blockIdx.x * blockDim.x + threadIdx.x;
  int wave = gtid >> 5;
  int lane = gtid & 31;
  int rowTiles = (N + 15) >> 4;
  if (wave >= rowTiles * colTiles) return;       // wave-uniform: EXEC stays full
  int rowTile = wave / colTiles;
  int colTile = wave - rowTile * colTiles;
  int l16 = lane & 15;
  int g   = lane >> 4;                           // half-wave group

  int aRow = rowTile * 16 + l16;
  if (aRow >= N) aRow = N - 1;                   // N is a multiple of 16 here
  const _Float16* aPtr = A + (size_t)aRow * KDIM;
  const _Float16* wPtr = W + (size_t)(colTile * 16 + l16) * KDIM;

  v8f acc = {};
#pragma unroll
  for (int k0 = 0; k0 < KDIM; k0 += 32) {
    // A fragment: lane holds row M=l16; halves h -> K = k0 + g*16 + h
    // (same contiguous K order as the B fragment)
    v8h a0 = *(const v8h*)(aPtr + k0 + g * 16);
    v8h a1 = *(const v8h*)(aPtr + k0 + g * 16 + 8);
    // B fragment: lane holds column N=l16; halves h -> K = k0 + g*16 + h
    v16h bv = *(const v16h*)(wPtr + k0 + g * 16);
    v16h av;
#pragma unroll
    for (int i = 0; i < 8; ++i) { av[i] = a0[i]; av[i + 8] = a1[i]; }
    acc = __builtin_amdgcn_wmma_f32_16x16x32_f16(false, av, false, bv,
                                                 (short)0, acc, false, false);
  }

  int col = colTile * 16 + l16;
  float b = bias[col];
  // D layout: element r of acc -> row = rowTile*16 + g*8 + r
  float* oPtr = out + (size_t)(rowTile * 16 + g * 8) * OUTD + col;
  if (rowTile * 16 + 16 <= N) {                  // full tile: branchless stores
#pragma unroll
    for (int r = 0; r < 8; ++r) oPtr[r * OUTD] = acc[r] + b;
  } else {                                       // tail tile (unused when N%16==0)
#pragma unroll
    for (int r = 0; r < 8; ++r) {
      if (rowTile * 16 + g * 8 + r < N) oPtr[r * OUTD] = acc[r] + b;
    }
  }
}

// ---- per-layer init: acc = bias, m = -inf (ordered 0), denom = 0 -----------

__global__ void init_layer_kernel(unsigned* __restrict__ mU,
                                  float* __restrict__ denom,
                                  float* __restrict__ acc,
                                  const float* __restrict__ bias, int N) {
  int i = blockIdx.x * blockDim.x + threadIdx.x;
  if (i < N * HID) acc[i] = bias[i & (HID - 1)];
  if (i < N * HEADS) { mU[i] = 0u; denom[i] = 0.0f; }
}

// ---- edge logits + segment max (one wave32 per edge) -----------------------

__global__ void edge_logits_kernel(const int* __restrict__ srcA,
                                   const int* __restrict__ dstA,
                                   int E, int N,
                                   const float* __restrict__ XL,
                                   const float* __restrict__ XR,
                                   const float* __restrict__ att,   // [4*32]
                                   float* __restrict__ logits,      // [(E+N)*4]
                                   unsigned* __restrict__ mU) {     // [N*4]
  int gtid = blockIdx.x * blockDim.x + threadIdx.x;
  int e    = gtid >> 5;
  int lane = gtid & 31;
  int total = E + N;
  if (e >= total) return;                         // wave-uniform
  int s = (e < E) ? srcA[e] : (e - E);            // self-loops appended
  int d = (e < E) ? dstA[e] : (e - E);
  const float* xl = XL + (size_t)s * HID;
  const float* xr = XR + (size_t)d * HID;
  float lg[HEADS];
#pragma unroll
  for (int h = 0; h < HEADS; ++h) {
    float v = xl[h * DHEAD + lane] + xr[h * DHEAD + lane];
    v = (v > 0.0f) ? v : v * NEG_SLOPE;
    v *= att[h * DHEAD + lane];
#pragma unroll
    for (int off = 16; off; off >>= 1) v += __shfl_xor(v, off, 32);
    lg[h] = v;                                    // butterfly: all lanes hold sum
  }
  if (lane < HEADS) {                             // lanes 0..3 write coalesced
    float v = lg[0];
    if (lane == 1) v = lg[1];
    if (lane == 2) v = lg[2];
    if (lane == 3) v = lg[3];
    logits[(size_t)e * HEADS + lane] = v;
    atomicMax(&mU[(size_t)d * HEADS + lane], orderedOf(v));
  }
}

// ---- exp + denominator (one thread per (edge, head)) -----------------------

__global__ void edge_exp_kernel(const int* __restrict__ dstA, int E, int N,
                                const unsigned* __restrict__ mU,
                                float* __restrict__ logits,      // in-place -> p
                                float* __restrict__ denom) {
  int idx = blockIdx.x * blockDim.x + threadIdx.x;
  int total = (E + N) * HEADS;
  if (idx >= total) return;
  int e = idx >> 2, h = idx & 3;
  int d = (e < E) ? dstA[e] : (e - E);
  float m = unorderedOf(mU[(size_t)d * HEADS + h]);
  float p = expf(logits[idx] - m);
  logits[idx] = p;
  atomicAdd(&denom[(size_t)d * HEADS + h], p);
}

// ---- weighted scatter-add (one wave32 per edge) ----------------------------

__global__ void edge_scatter_kernel(const int* __restrict__ srcA,
                                    const int* __restrict__ dstA,
                                    int E, int N,
                                    const float* __restrict__ XL,
                                    const float* __restrict__ p,      // [(E+N)*4]
                                    const float* __restrict__ denom,  // [N*4]
                                    float* __restrict__ acc) {        // [N*128]
  int gtid = blockIdx.x * blockDim.x + threadIdx.x;
  int e    = gtid >> 5;
  int lane = gtid & 31;
  int total = E + N;
  if (e >= total) return;
  int s = (e < E) ? srcA[e] : (e - E);
  int d = (e < E) ? dstA[e] : (e - E);
  const float* xl = XL + (size_t)s * HID;
  float* o = acc + (size_t)d * HID;
  // lanes 0..3 load per-head alpha once; broadcast to the wave via shuffles
  float alpha = 0.0f;
  if (lane < HEADS) {
    float den = denom[(size_t)d * HEADS + lane];
    alpha = p[(size_t)e * HEADS + lane] / fmaxf(den, 1e-16f);
  }
  float a0 = __shfl(alpha, 0, 32);
  float a1 = __shfl(alpha, 1, 32);
  float a2 = __shfl(alpha, 2, 32);
  float a3 = __shfl(alpha, 3, 32);
  atomicAdd(&o[0 * DHEAD + lane], a0 * xl[0 * DHEAD + lane]);
  atomicAdd(&o[1 * DHEAD + lane], a1 * xl[1 * DHEAD + lane]);
  atomicAdd(&o[2 * DHEAD + lane], a2 * xl[2 * DHEAD + lane]);
  atomicAdd(&o[3 * DHEAD + lane], a3 * xl[3 * DHEAD + lane]);
}

// ---------------------------------------------------------------------------

extern "C" void kernel_launch(void* const* d_in, const int* in_sizes, int n_in,
                              void* d_out, int out_size, void* d_ws, size_t ws_size,
                              hipStream_t stream) {
  const float* x    = (const float*)d_in[0];
  const int*   ei   = (const int*)d_in[1];        // int32 (JAX default x64 off)
  const float* Wl0  = (const float*)d_in[2];
  const float* bl0  = (const float*)d_in[3];
  const float* Wr0  = (const float*)d_in[4];
  const float* br0  = (const float*)d_in[5];
  const float* att0 = (const float*)d_in[6];
  const float* bias0= (const float*)d_in[7];
  const float* Wl1  = (const float*)d_in[8];
  const float* bl1  = (const float*)d_in[9];
  const float* Wr1  = (const float*)d_in[10];
  const float* br1  = (const float*)d_in[11];
  const float* att1 = (const float*)d_in[12];
  const float* bias1= (const float*)d_in[13];
  const float* dW   = (const float*)d_in[14];
  const float* db   = (const float*)d_in[15];

  const int N = in_sizes[0] / HID;                // 50000 (multiple of 16)
  const int E = in_sizes[1] / 2;                  // 600000
  const int* srcA = ei;                           // edge_index[0]
  const int* dstA = ei + E;                       // edge_index[1]

  // ---- workspace carve-up (all 256B aligned) ----
  char* wp = (char*)d_ws;
  auto alloc = [&](size_t bytes) -> void* {
    void* q = (void*)wp;
    wp += (bytes + 255) & ~(size_t)255;
    return q;
  };
  float*     XL     = (float*)    alloc((size_t)N * HID * 4);
  float*     XR     = (float*)    alloc((size_t)N * HID * 4);
  _Float16*  H16    = (_Float16*) alloc((size_t)N * HID * 2);
  _Float16*  W16a   = (_Float16*) alloc((size_t)HID * KDIM * 2);
  _Float16*  W16b   = (_Float16*) alloc((size_t)HID * KDIM * 2);
  _Float16*  W16d   = (_Float16*) alloc((size_t)DHEAD * KDIM * 2);
  float*     logits = (float*)    alloc((size_t)(E + N) * HEADS * 4);
  unsigned*  mU     = (unsigned*) alloc((size_t)N * HEADS * 4);
  float*     denom  = (float*)    alloc((size_t)N * HEADS * 4);
  float*     acc    = (float*)    alloc((size_t)N * HID * 4);

  const int BLK = 256;
  auto blocks = [](long long n, int b) { return (int)((n + b - 1) / b); };

  const int nodeElems  = N * HID;                       // 6.4M
  const int totalE     = E + N;                         // edges + self loops
  const int gemmWaves  = ((N + 15) / 16) * (HID / 16);  // 25000
  const int gemmBlocks = blocks((long long)gemmWaves * 32, BLK);
  const int downWaves  = ((N + 15) / 16) * (DHEAD / 16);
  const int downBlocks = blocks((long long)downWaves * 32, BLK);
  const int edgeBlocks = blocks((long long)totalE * 32, BLK);
  const int ehBlocks   = blocks((long long)totalE * HEADS, BLK);
  const int initBlocks = blocks(nodeElems, BLK);
  const int cvtNBlocks = blocks(nodeElems, BLK);
  const int cvtWBlocks = blocks(HID * KDIM, BLK);

  // ================= layer 0 =================
  cvt_f32_to_f16_kernel<<<cvtNBlocks, BLK, 0, stream>>>(x,   H16,  nodeElems);
  cvt_f32_to_f16_kernel<<<cvtWBlocks, BLK, 0, stream>>>(Wl0, W16a, HID * KDIM);
  cvt_f32_to_f16_kernel<<<cvtWBlocks, BLK, 0, stream>>>(Wr0, W16b, HID * KDIM);
  wmma_gemm_bias_kernel<HID><<<gemmBlocks, BLK, 0, stream>>>(H16, W16a, bl0, XL, N);
  wmma_gemm_bias_kernel<HID><<<gemmBlocks, BLK, 0, stream>>>(H16, W16b, br0, XR, N);
  init_layer_kernel<<<initBlocks, BLK, 0, stream>>>(mU, denom, acc, bias0, N);
  edge_logits_kernel<<<edgeBlocks, BLK, 0, stream>>>(srcA, dstA, E, N, XL, XR, att0, logits, mU);
  edge_exp_kernel<<<ehBlocks, BLK, 0, stream>>>(dstA, E, N, mU, logits, denom);
  edge_scatter_kernel<<<edgeBlocks, BLK, 0, stream>>>(srcA, dstA, E, N, XL, logits, denom, acc);

  // ================= layer 1 ================= (input = acc, consumed before re-init)
  cvt_f32_to_f16_kernel<<<cvtNBlocks, BLK, 0, stream>>>(acc, H16,  nodeElems);
  cvt_f32_to_f16_kernel<<<cvtWBlocks, BLK, 0, stream>>>(Wl1, W16a, HID * KDIM);
  cvt_f32_to_f16_kernel<<<cvtWBlocks, BLK, 0, stream>>>(Wr1, W16b, HID * KDIM);
  wmma_gemm_bias_kernel<HID><<<gemmBlocks, BLK, 0, stream>>>(H16, W16a, bl1, XL, N);
  wmma_gemm_bias_kernel<HID><<<gemmBlocks, BLK, 0, stream>>>(H16, W16b, br1, XR, N);
  init_layer_kernel<<<initBlocks, BLK, 0, stream>>>(mU, denom, acc, bias1, N);
  edge_logits_kernel<<<edgeBlocks, BLK, 0, stream>>>(srcA, dstA, E, N, XL, XR, att1, logits, mU);
  edge_exp_kernel<<<ehBlocks, BLK, 0, stream>>>(dstA, E, N, mU, logits, denom);
  edge_scatter_kernel<<<edgeBlocks, BLK, 0, stream>>>(srcA, dstA, E, N, XL, logits, denom, acc);

  // ================= down projection =================
  cvt_f32_to_f16_kernel<<<cvtNBlocks, BLK, 0, stream>>>(acc, H16, nodeElems);
  cvt_f32_to_f16_kernel<<<blocks(DHEAD * KDIM, BLK), BLK, 0, stream>>>(dW, W16d, DHEAD * KDIM);
  wmma_gemm_bias_kernel<DHEAD><<<downBlocks, BLK, 0, stream>>>(H16, W16d, db, (float*)d_out, N);
}
